// graph_smooth_43284680409688
// MI455X (gfx1250) — matrix operand order repaired and checked by reading, not compiled, not run on hardware
//
#include <hip/hip_runtime.h>
#include <hip/hip_bf16.h>
#include <stdint.h>

// Sizes fixed by the reference problem.
#define NN 4096   // nodes
#define DE 128    // embedding dim
#define DO 64     // output dim

#define CHUNK_K 128            // K elements staged per LDS buffer
#define BROW   (CHUNK_K + 8)   // LDS row pitch in bf16 (16B pad -> bank rotation)

typedef __attribute__((ext_vector_type(16))) __bf16 v16bf;
typedef __attribute__((ext_vector_type(8)))  float  v8f;
typedef unsigned int u32x4 __attribute__((ext_vector_type(4)));
typedef int          i32x8 __attribute__((ext_vector_type(8)));
typedef int          i32x4 __attribute__((ext_vector_type(4)));

// float -> bf16 bits, round-to-nearest-even (pure integer).
__device__ __forceinline__ uint16_t f2bf(float f) {
  uint32_t u = __float_as_uint(f);
  uint32_t r = 0x7FFFu + ((u >> 16) & 1u);
  return (uint16_t)((u + r) >> 16);
}

// A-fragment (16x32 bf16, M x K) per ISA 7.12.2, from row-major global memory:
//   lanes 0-15 : row M=lane,    K = {k0+0..7, k0+16..23}
//   lanes 16-31: row M=lane-16, K = {k0+8..15, k0+24..31}
__device__ __forceinline__ v16bf load_a_frag(const uint16_t* __restrict__ M,
                                             int ld, int row0, int k0) {
  const int lane = threadIdx.x & 31;
  const uint16_t* p = M + (size_t)(row0 + (lane & 15)) * ld
                        + (k0 + ((lane & 16) ? 8 : 0));
  union { v16bf v; uint4 u[2]; } r;
  r.u[0] = *(const uint4*)(p);
  r.u[1] = *(const uint4*)(p + 16);
  return r.v;
}

// B-fragment (32x16 bf16, K x N) from the LDS-staged B^T chunk [64][BROW]:
//   lanes 0-15 : col N=lane,    K = {kk+0..7, kk+16..23}
//   lanes 16-31: col N=lane-16, K = {kk+8..15, kk+24..31}
__device__ __forceinline__ v16bf load_b_frag_lds(const uint16_t* sb,
                                                 int col0, int kk) {
  const int lane = threadIdx.x & 31;
  const uint16_t* p = sb + (size_t)(col0 + (lane & 15)) * BROW
                         + (kk + ((lane & 16) ? 8 : 0));
  union { v16bf v; uint4 u[2]; } r;
  r.u[0] = *(const uint4*)(p);
  r.u[1] = *(const uint4*)(p + 16);
  return r.v;
}

// TDM: one tensor_load_to_lds moves the whole 64 x CHUNK_K bf16 strip of B^T
// into LDS (tracked by TENSORcnt). The TDM pad feature inserts 16B after every
// 256B row, reproducing the BROW bank-rotation pitch.
__device__ __forceinline__ void tdm_fill_b(const uint16_t* __restrict__ BT,
                                           int K, uint16_t* shbuf, int kbase) {
  const uint64_t ga  = (uint64_t)(uintptr_t)(BT + kbase);      // tile start
  const uint32_t lds = (uint32_t)(uintptr_t)shbuf;             // low32 = LDS byte offset

  u32x4 g0;
  g0.x = 1u;                                        // count = 1 (valid descriptor)
  g0.y = lds;                                       // lds_addr
  g0.z = (uint32_t)ga;                              // global_addr[31:0]
  g0.w = (uint32_t)((ga >> 32) & 0x01FFFFFFu)       // global_addr[56:32]
       | 0x80000000u;                               // type = 2 ("image")

  i32x8 g1;
  g1[0] = (1 << 16)                                 // data_size = 2 bytes
        | (1 << 20)                                 // pad_enable (load)
        | (5 << 22)                                 // pad_interval: 64 DWORDs = 256B
        | (3 << 25);                                // pad_amount: 4 DWORDs = 16B
  g1[1] = (CHUNK_K & 0xFFFF) << 16;                 // tensor_dim0[15:0] (= tile, no OOB)
  g1[2] = (DO << 16);                               // tensor_dim0[31:16]=0, tensor_dim1=64
  g1[3] = (CHUNK_K << 16);                          // tensor_dim1[31:16]=0, tile_dim0=128
  g1[4] = DO;                                       // tile_dim1=64, tile_dim2=0 (2D)
  g1[5] = K;                                        // tensor_dim0_stride[31:0]
  g1[6] = 0;                                        // stride[47:32]=0, dim1_stride lo=0
  g1[7] = 0;

  // 2D tile (tile_dim2=0): all higher-dim descriptor words are don't-care.
  i32x4 gz4 = {0, 0, 0, 0};
  i32x8 gz8 = {0, 0, 0, 0, 0, 0, 0, 0};
  // amdgpu-toolchain (clang-23) 6-arg form:
  // (u32x4 g0, i32x8 g1, i32x4, i32x4, i32x8, i32 cpol)
  __builtin_amdgcn_tensor_load_to_lds(g0, g1, gz4, gz4, gz8, 0);
}

// C[M x 64] (f32) = A_bf16[M x K, row-major] @ B (given as B^T bf16 [64 x K]).
// 8 waves/block; each wave owns a 16-row block and all 4 column tiles (A-frag
// reused 4x). B^T is double-buffered in LDS by the Tensor Data Mover (wave 0
// issues the DMA; TENSORcnt + workgroup barrier publish completion). A streams
// from global with prefetch. K must be a multiple of CHUNK_K.
__global__ void __launch_bounds__(256)
wmma_gemm_bf16(const uint16_t* __restrict__ A, const uint16_t* __restrict__ BT,
               float* __restrict__ C, int K) {
  __shared__ __align__(16) uint16_t shB[2][DO * BROW];
  const int wave = threadIdx.x >> 5;
  const int lane = threadIdx.x & 31;
  const int m0   = (blockIdx.x * 8 + wave) * 16;
  const int nchunks = K / CHUNK_K;
  const bool leader = (wave == 0);   // tensor ops issue per wave: exactly one issues

  if (leader) tdm_fill_b(BT, K, shB[0], 0);

  v8f a0 = {}, a1 = {}, a2 = {}, a3 = {};
  for (int c = 0; c < nchunks; ++c) {
    uint16_t* sb = shB[c & 1];
    if (leader) {
      if (c + 1 < nchunks) {
        tdm_fill_b(BT, K, shB[(c + 1) & 1], (c + 1) * CHUNK_K);
        __builtin_amdgcn_s_wait_tensorcnt(1);   // in-order: current buffer done
      } else {
        __builtin_amdgcn_s_wait_tensorcnt(0);
      }
    }
    __syncthreads();  // TDM writes to sb visible to all waves

#pragma unroll
    for (int kk = 0; kk < CHUNK_K; kk += 32) {
      const int k0 = c * CHUNK_K + kk;
      if (k0 + 512 < K)  // keep A stream ahead (global_prefetch_b8)
        __builtin_prefetch(A + (size_t)(m0 + (threadIdx.x & 15)) * K + k0 + 512, 0, 1);
      v16bf af = load_a_frag(A, K, m0, k0);
      v16bf b0 = load_b_frag_lds(sb,  0, kk);
      v16bf b1 = load_b_frag_lds(sb, 16, kk);
      v16bf b2 = load_b_frag_lds(sb, 32, kk);
      v16bf b3 = load_b_frag_lds(sb, 48, kk);
      a0 = __builtin_amdgcn_wmma_f32_16x16x32_bf16(false, af, false, b0, (short)0, a0, false, false);
      a1 = __builtin_amdgcn_wmma_f32_16x16x32_bf16(false, af, false, b1, (short)0, a1, false, false);
      a2 = __builtin_amdgcn_wmma_f32_16x16x32_bf16(false, af, false, b2, (short)0, a2, false, false);
      a3 = __builtin_amdgcn_wmma_f32_16x16x32_bf16(false, af, false, b3, (short)0, a3, false, false);
    }
    __syncthreads();  // done reading sb before the TDM refills it next-next chunk
  }

  // C/D layout: lane 0-15 -> N=lane, M=vgpr; lane 16-31 -> N=lane-16, M=vgpr+8.
  float* Cr = C + (size_t)(m0 + ((lane & 16) ? 8 : 0)) * DO + (lane & 15);
#pragma unroll
  for (int i = 0; i < 8; ++i) {
    Cr[(size_t)i * DO +  0] = a0[i];
    Cr[(size_t)i * DO + 16] = a1[i];
    Cr[(size_t)i * DO + 32] = a2[i];
    Cr[(size_t)i * DO + 48] = a3[i];
  }
}

// Zero a bf16 buffer, 16 bytes per store (n16 = count of uint4 chunks).
__global__ void zero16_kernel(uint16_t* __restrict__ p, size_t n16) {
  size_t i = (size_t)blockIdx.x * blockDim.x + threadIdx.x;
  const size_t stride = (size_t)gridDim.x * blockDim.x;
  uint4 z; z.x = z.y = z.z = z.w = 0u;
  uint4* q = (uint4*)p;
  for (; i < n16; i += stride) q[i] = z;
}

// A[src, dst] = bf16(w)  (duplicate-edge race -> arbitrary winner, matching
// the reference's unspecified last-write-wins scatter).
__global__ void scatter_edges_kernel(const int* __restrict__ ei,
                                     const float* __restrict__ w,
                                     uint16_t* __restrict__ A, int E) {
  int e = blockIdx.x * blockDim.x + threadIdx.x;
  if (e < E) {
    int s = ei[e];
    int d = ei[E + e];
    A[(size_t)s * NN + d] = f2bf(w[e]);
  }
}

// y[i] = bf16(x[i])
__global__ void cvt_f32_bf16_kernel(const float* __restrict__ x,
                                    uint16_t* __restrict__ y, int n) {
  int i = blockIdx.x * blockDim.x + threadIdx.x;
  const int stride = gridDim.x * blockDim.x;
  for (; i < n; i += stride) y[i] = f2bf(x[i]);
}

// XT[c, r] = bf16(X[r, c]), X is R x C row-major f32.
__global__ void transpose_f32_bf16_kernel(const float* __restrict__ X,
                                          uint16_t* __restrict__ XT,
                                          int R, int C) {
  int i = blockIdx.x * blockDim.x + threadIdx.x;
  const int stride = gridDim.x * blockDim.x;
  const int n = R * C;
  for (; i < n; i += stride) {
    int r = i / C, c = i - r * C;
    XT[(size_t)c * R + r] = f2bf(X[i]);
  }
}

extern "C" void kernel_launch(void* const* d_in, const int* in_sizes, int n_in,
                              void* d_out, int out_size, void* d_ws, size_t ws_size,
                              hipStream_t stream) {
  const float* node_features = (const float*)d_in[0];  // [4096,128] f32
  const float* Emb           = (const float*)d_in[1];  // [128,64]  f32
  const int*   edge_index    = (const int*)d_in[2];    // [2,E]     i32
  const float* edge_weights  = (const float*)d_in[3];  // [E]       f32
  const int E = in_sizes[3];
  float* out = (float*)d_out;                          // [4096,64] f32

  // Carve workspace (all offsets 16B-aligned).
  char* ws = (char*)d_ws;
  uint16_t* Abf  = (uint16_t*)ws; ws += (size_t)NN * NN * 2;  // 32 MB bf16 adjacency
  uint16_t* nfb  = (uint16_t*)ws; ws += (size_t)NN * DE * 2;  // bf16 node features
  uint16_t* EmbT = (uint16_t*)ws; ws += (size_t)DO * DE * 2;  // bf16 Emb^T [64,128]
  uint16_t* xT   = (uint16_t*)ws; ws += (size_t)DO * NN * 2;  // bf16 x^T  [64,4096]
  float*    x    = (float*)ws;    ws += (size_t)NN * DO * 4;  // f32 x     [4096,64]

  // 1) Build bf16 dense adjacency.
  zero16_kernel<<<4096, 256, 0, stream>>>(Abf, (size_t)NN * NN / 8);
  scatter_edges_kernel<<<(E + 255) / 256, 256, 0, stream>>>(edge_index, edge_weights, Abf, E);

  // 2) Projection x = node_features @ Emb via WMMA (K=128 -> one TDM chunk).
  cvt_f32_bf16_kernel<<<1024, 256, 0, stream>>>(node_features, nfb, NN * DE);
  transpose_f32_bf16_kernel<<<32, 256, 0, stream>>>(Emb, EmbT, DE, DO);
  wmma_gemm_bf16<<<NN / 128, 256, 0, stream>>>(nfb, EmbT, x, DE);

  // 3) x = A @ x, three times (K=4096). Snapshot x into bf16 x^T each pass,
  //    so the f32 x buffer can be overwritten in place; last pass -> d_out.
  for (int it = 0; it < 3; ++it) {
    transpose_f32_bf16_kernel<<<1024, 256, 0, stream>>>(x, xT, NN, DO);
    wmma_gemm_bf16<<<NN / 128, 256, 0, stream>>>(Abf, xT, (it == 2) ? out : x, NN);
  }
}
